// DCN_l1_75539884802538
// MI455X (gfx1250) — matrix operand order repaired and checked by reading, not compiled, not run on hardware
//
#include <hip/hip_runtime.h>

// ---------------------------------------------------------------------------
// MI455X (gfx1250) WMMA implementation of the PCD/DCN alignment block.
// All conv reductions run through v_wmma_f32_16x16x32_bf16; conv halo tiles
// are staged into LDS with GLOBAL_LOAD_ASYNC_TO_LDS_B128 when available.
// Intermediate activations are bf16 NHWC (channel-last) so LDS staging is a
// raw byte copy and epilogue stores are coalesced.
// ---------------------------------------------------------------------------

typedef __attribute__((ext_vector_type(16))) __bf16 v16bf;
typedef __attribute__((ext_vector_type(8)))  float  v8f;
typedef int v4i __attribute__((vector_size(16)));

struct alignas(16) Q16 { unsigned int a, b, c, d; };   // 16-byte raw chunk

#define HH 128
#define WW 128
#define NB 4
#define NFC 64

__device__ inline float  bf2f(__bf16 v) { return (float)v; }
__device__ inline __bf16 f2bf(float v)  { return (__bf16)v; }

#if defined(__HIP_DEVICE_COMPILE__) &&                                        \
    __has_builtin(__builtin_amdgcn_global_load_async_to_lds_b128) &&          \
    __has_builtin(__builtin_amdgcn_s_wait_asynccnt)
#define HAVE_ASYNC_LDS 1
#else
#define HAVE_ASYNC_LDS 0
#endif

#if HAVE_ASYNC_LDS
__device__ inline void async_copy16(const void* gsrc, void* ldst) {
  __builtin_amdgcn_global_load_async_to_lds_b128(
      (__attribute__((address_space(1))) v4i*)(void*)gsrc,
      (__attribute__((address_space(3))) v4i*)ldst, 0, 0);
}
#endif

// ---------------------------------------------------------------------------
// Pack conv weights [Cout][Cin][3][3] (f32) into WMMA B-fragment order (bf16).
// K ordering is TAP-MAJOR: k = tap*Cin + ic, so 8-aligned runs of k are 8
// consecutive input channels (matches the A-fragment b128 loads).
// Fragment element (nt,kc,lane,e) holds W[oc = nt*16 + (lane&15)]
// at k = kc*32 + (e>>3)*16 + (lane>>4)*8 + (e&7)   (matches A fragment order,
// N = lane&15).
// ---------------------------------------------------------------------------
__global__ void prepack_w(const float* __restrict__ w, __bf16* __restrict__ wp,
                          int Cin, int Cout, int NT, int KC) {
  int idx = blockIdx.x * blockDim.x + threadIdx.x;
  int total = NT * KC * 512;
  if (idx >= total) return;
  int e    = idx & 15;
  int lane = (idx >> 4) & 31;
  int kc   = (idx >> 9) % KC;
  int nt   = idx / (512 * KC);
  int k    = kc * 32 + ((e >> 3) << 4) + ((lane >> 4) << 3) + (e & 7);     // k < Cin*9 by construction
  int oc   = nt * 16 + (lane & 15);
  int tap  = k / Cin;
  int ic   = k % Cin;
  float v  = (oc < Cout) ? w[(oc * Cin + ic) * 9 + tap] : 0.0f;
  wp[idx] = f2bf(v);
}

// NCHW f32 -> NHWC bf16 (channels-last) copy of nbr_feat for the DCN gather.
__global__ void transpose_bf16(const float* __restrict__ x, __bf16* __restrict__ xt) {
  int idx = blockIdx.x * blockDim.x + threadIdx.x;      // ((b*H+y)*W+x)*64 + c
  if (idx >= NB * HH * WW * NFC) return;
  int c  = idx & (NFC - 1);
  int p  = idx >> 6;
  int xw = p & (WW - 1); p >>= 7;
  int yv = p & (HH - 1); p >>= 7;
  int b  = p;
  xt[idx] = f2bf(x[((b * NFC + c) * HH + yv) * WW + xw]);
}

// ---------------------------------------------------------------------------
// Generic 3x3 conv (pad 1) as implicit GEMM on WMMA bf16 16x16x32.
// Block = 128 threads (4 waves). Block tile: one row y, 64 x-positions.
// Wave w owns positions x0+16w .. x0+16w+15 (the GEMM M=16 dimension).
// K = Cin*9 reduced in chunks of 32; N = Cout in NT tiles of 16.
// Input halo (3 rows x 66 cols x Cin) staged in LDS as bf16, channel-padded
// to Cin+8 so the 16 A-lanes hit distinct bank groups.
//   CONCAT=true : input = concat([in0,in1]) f32 NCHW (VGPR staging + convert)
//   CONCAT=false: input = in0 bf16 NHWC (async b128 LDS staging)
// ---------------------------------------------------------------------------
template <int CIN, int COUT, int NT, bool CONCAT, bool OFF_EP>
__global__ __launch_bounds__(128) void conv3x3_wmma(
    const void* __restrict__ in0, const void* __restrict__ in1,
    const __bf16* __restrict__ wpack, const float* __restrict__ bias,
    __bf16* __restrict__ out_bf, float* __restrict__ out_off,
    float* __restrict__ out_mask) {
  constexpr int KC   = (CIN * 9) / 32;
  constexpr int CINP = CIN + 8;
  __shared__ __align__(16) __bf16 smem[3 * 66 * CINP];

  const int x0  = blockIdx.x * 64;
  const int y   = blockIdx.y;
  const int b   = blockIdx.z;
  const int tid = threadIdx.x;

  if constexpr (CONCAT) {
    // f32 NCHW x2 -> bf16 LDS, element-wise (conversion required)
    for (int idx = tid; idx < 3 * 66 * CIN; idx += 128) {
      int kh = idx / (66 * CIN);
      int r  = idx - kh * (66 * CIN);
      int xx = r / CIN;
      int ic = r - xx * CIN;
      int gy = y + kh - 1;
      int gx = x0 - 1 + xx;
      float v = 0.0f;
      if (gy >= 0 && gy < HH && gx >= 0 && gx < WW) {
        if (ic < CIN / 2)
          v = ((const float*)in0)[((b * (CIN / 2) + ic) * HH + gy) * WW + gx];
        else
          v = ((const float*)in1)[((b * (CIN / 2) + (ic - CIN / 2)) * HH + gy) * WW + gx];
      }
      smem[(kh * 66 + xx) * CINP + ic] = f2bf(v);
    }
  } else {
    // bf16 NHWC -> LDS: pure 16-byte chunk copies (8 channels per chunk).
    constexpr int NCH = CIN / 8;
    for (int idx = tid; idx < 3 * 66 * NCH; idx += 128) {
      int kh = idx / (66 * NCH);
      int r  = idx - kh * (66 * NCH);
      int xx = r / NCH;
      int g8 = r - xx * NCH;
      int gy = y + kh - 1;
      int gx = x0 - 1 + xx;
      __bf16* dst = &smem[(kh * 66 + xx) * CINP + g8 * 8];
      if (gy >= 0 && gy < HH && gx >= 0 && gx < WW) {
        const __bf16* src =
            ((const __bf16*)in0) + (((size_t)(b * HH + gy) * WW + gx) * CIN + g8 * 8);
#if HAVE_ASYNC_LDS
        async_copy16(src, dst);           // global_load_async_to_lds_b128
#else
        *(Q16*)dst = *(const Q16*)src;
#endif
      } else {
        Q16 z; z.a = 0; z.b = 0; z.c = 0; z.d = 0;
        *(Q16*)dst = z;
      }
    }
#if HAVE_ASYNC_LDS
    __builtin_amdgcn_s_wait_asynccnt(0);
#endif
  }
  __syncthreads();

  const int wv   = tid >> 5;
  const int lane = tid & 31;
  const int half = lane >> 4;   // A/B lane K-half
  const int mrow = lane & 15;   // GEMM M row (position within 16)

  const v8f vzero = {0.f, 0.f, 0.f, 0.f, 0.f, 0.f, 0.f, 0.f};
  v8f acc[NT];
#pragma unroll
  for (int nt = 0; nt < NT; ++nt) acc[nt] = vzero;

  for (int kc = 0; kc < KC; ++kc) {
    // A fragment 16x32 bf16: element e -> k = (e/8)*16 + half*8 + e%8.
    // Each 8-run = 8 consecutive channels at one (kh, xx): one ds_load_b128.
    v16bf afrag;
#pragma unroll
    for (int g = 0; g < 2; ++g) {
      int k   = kc * 32 + g * 16 + half * 8;
      int tap = k / CIN;           // tap-major K
      int ic0 = k % CIN;           // multiple of 8
      int kh  = tap / 3;
      int kw  = tap - kh * 3;
      int xx  = 16 * wv + mrow + kw;
      ((Q16*)&afrag)[g] = *(const Q16*)&smem[(kh * 66 + xx) * CINP + ic0];
    }
#pragma unroll
    for (int nt = 0; nt < NT; ++nt) {
      v16bf bfrag;  // prepacked: lane-contiguous 32B -> two b128 global loads
      const Q16* wp = (const Q16*)(wpack + ((size_t)(nt * KC + kc) * 32 + lane) * 16);
      ((Q16*)&bfrag)[0] = wp[0];
      ((Q16*)&bfrag)[1] = wp[1];
      acc[nt] = __builtin_amdgcn_wmma_f32_16x16x32_bf16(
          false, afrag, false, bfrag, (short)0, acc[nt], false, false);
    }
  }

  // ---- epilogue: D element r of lane L -> M = r + 8*(L>=16), N = L&15
#pragma unroll
  for (int nt = 0; nt < NT; ++nt) {
    int oc = nt * 16 + (lane & 15);
    float bv = (oc < COUT) ? bias[oc] : 0.0f;
#pragma unroll
    for (int r = 0; r < 8; ++r) {
      int xp = x0 + 16 * wv + r + 8 * half;
      float v = acc[nt][r] + bv;
      if constexpr (OFF_EP) {
        if (oc < 18)        // offset = 15*tanh(conv_out[0:18]), f32 NCHW
          out_off[((b * 18 + oc) * HH + y) * WW + xp] = 15.0f * tanhf(v);
        else if (oc < 27)   // mask = sigmoid(conv_out[18:27]), f32 NCHW
          out_mask[((b * 9 + (oc - 18)) * HH + y) * WW + xp] =
              1.0f / (1.0f + expf(-v));
      } else {              // lrelu -> bf16 NHWC activation (coalesced)
        out_bf[((size_t)(b * HH + y) * WW + xp) * COUT + oc] =
            f2bf(v >= 0.0f ? v : 0.1f * v);
      }
    }
  }
}

// ---------------------------------------------------------------------------
// Modulated deformable conv as WMMA GEMM: A[m][k] = mask * bilinear sample,
// B[k][oc] = prepacked w_dcn. Same tiling as conv3x3_wmma. Each A 8-run is
// one (position, tap) with 8 consecutive channels from the NHWC bf16 copy:
// 4 corner b128 loads + f32 bilinear combine + mask, then bf16 fragment.
// ---------------------------------------------------------------------------
__global__ __launch_bounds__(128) void dcn_wmma(
    const __bf16* __restrict__ nbrT, const float* __restrict__ offb,
    const float* __restrict__ maskb, const __bf16* __restrict__ wpd,
    const float* __restrict__ bias, float* __restrict__ feat) {
  constexpr int KC = 18;  // 64*9/32
  const int x0  = blockIdx.x * 64;
  const int y   = blockIdx.y;
  const int b   = blockIdx.z;
  const int tid = threadIdx.x;
  const int wv = tid >> 5, lane = tid & 31, half = lane >> 4, mrow = lane & 15;
  const int xpos = x0 + 16 * wv + mrow;

  const v8f vzero = {0.f, 0.f, 0.f, 0.f, 0.f, 0.f, 0.f, 0.f};
  v8f acc[4];
#pragma unroll
  for (int nt = 0; nt < 4; ++nt) acc[nt] = vzero;

  for (int kc = 0; kc < KC; ++kc) {
    v16bf afrag;
#pragma unroll
    for (int g = 0; g < 2; ++g) {
      int k   = kc * 32 + g * 16 + half * 8;
      int tap = k >> 6;         // k / 64, tap-major
      int ic0 = k & 63;         // multiple of 8
      float oy = offb[((b * 18 + 2 * tap) * HH + y) * WW + xpos];
      float ox = offb[((b * 18 + 2 * tap + 1) * HH + y) * WW + xpos];
      float mk = maskb[((b * 9 + tap) * HH + y) * WW + xpos];
      float py = (float)y + (float)(tap / 3 - 1) + oy;
      float px = (float)xpos + (float)(tap % 3 - 1) + ox;
      float fy = floorf(py), fx = floorf(px);
      int iy = (int)fy, ix = (int)fx;
      float wy = py - fy, wx = px - fx;
      float s[8];
#pragma unroll
      for (int i = 0; i < 8; ++i) s[i] = 0.0f;
#pragma unroll
      for (int dy = 0; dy < 2; ++dy) {
#pragma unroll
        for (int dx = 0; dx < 2; ++dx) {
          int yy = iy + dy, xc = ix + dx;
          if (yy >= 0 && yy < HH && xc >= 0 && xc < WW) {
            Q16 q = *(const Q16*)&nbrT[(((size_t)(b * HH + yy) * WW + xc) << 6) + ic0];
            const __bf16* pv = (const __bf16*)&q;
            float wgt = (dy ? wy : 1.0f - wy) * (dx ? wx : 1.0f - wx);
#pragma unroll
            for (int i = 0; i < 8; ++i) s[i] += wgt * bf2f(pv[i]);
          }
        }
      }
      __bf16* ah = ((__bf16*)&afrag) + g * 8;
#pragma unroll
      for (int i = 0; i < 8; ++i) ah[i] = f2bf(s[i] * mk);
    }
#pragma unroll
    for (int nt = 0; nt < 4; ++nt) {
      v16bf bfrag;
      const Q16* wp = (const Q16*)(wpd + ((size_t)(nt * KC + kc) * 32 + lane) * 16);
      ((Q16*)&bfrag)[0] = wp[0];
      ((Q16*)&bfrag)[1] = wp[1];
      acc[nt] = __builtin_amdgcn_wmma_f32_16x16x32_bf16(
          false, afrag, false, bfrag, (short)0, acc[nt], false, false);
    }
  }
#pragma unroll
  for (int nt = 0; nt < 4; ++nt) {
    int oc = nt * 16 + (lane & 15);
    float bv = bias[oc];
#pragma unroll
    for (int r = 0; r < 8; ++r) {
      int xp = x0 + 16 * wv + r + 8 * half;
      float v = acc[nt][r] + bv;
      feat[((b * NFC + oc) * HH + y) * WW + xp] = (v >= 0.0f ? v : 0.1f * v);
    }
  }
}

// ---------------------------------------------------------------------------
// Workspace: ~25.7 MB (packed weights + 2 bf16 NHWC ping-pong activations +
// NHWC bf16 copy of nbr_feat). d_out layout: feat | offset | mask (f32, flat).
// ---------------------------------------------------------------------------
extern "C" void kernel_launch(void* const* d_in, const int* in_sizes, int n_in,
                              void* d_out, int out_size, void* d_ws, size_t ws_size,
                              hipStream_t stream) {
  (void)in_sizes; (void)n_in; (void)out_size; (void)ws_size;
  const float* nbr  = (const float*)d_in[0];
  const float* ref  = (const float*)d_in[1];
  const float* w1   = (const float*)d_in[2];
  const float* b1   = (const float*)d_in[3];
  const float* w2   = (const float*)d_in[4];
  const float* b2   = (const float*)d_in[5];
  const float* w3   = (const float*)d_in[6];
  const float* b3   = (const float*)d_in[7];
  const float* woff = (const float*)d_in[8];
  const float* boff = (const float*)d_in[9];
  const float* wdcn = (const float*)d_in[10];
  const float* bdcn = (const float*)d_in[11];

  float* out = (float*)d_out;
  const size_t FEAT = (size_t)NB * NFC * HH * WW;  // 4194304
  const size_t OFFS = (size_t)NB * 18 * HH * WW;   // 1179648
  float* out_feat = out;
  float* out_off  = out + FEAT;
  float* out_mask = out + FEAT + OFFS;

  char* ws = (char*)d_ws;
  size_t off = 0;
  auto carve = [&](size_t bytes) {
    char* p = ws + off;
    off = (off + bytes + 255) & ~(size_t)255;
    return p;
  };
  __bf16* wp1  = (__bf16*)carve((size_t)4 * 36 * 512 * 2);
  __bf16* wp2  = (__bf16*)carve((size_t)4 * 18 * 512 * 2);
  __bf16* wp3  = (__bf16*)carve((size_t)4 * 18 * 512 * 2);
  __bf16* wpo  = (__bf16*)carve((size_t)2 * 18 * 512 * 2);
  __bf16* wpd  = (__bf16*)carve((size_t)4 * 18 * 512 * 2);
  __bf16* actA = (__bf16*)carve(FEAT * 2);
  __bf16* actB = (__bf16*)carve(FEAT * 2);
  __bf16* nbrT = (__bf16*)carve(FEAT * 2);

  // weight prepack (fragment order, bf16) + NHWC bf16 copy of nbr
  prepack_w<<<(4 * 36 * 512 + 255) / 256, 256, 0, stream>>>(w1, wp1, 128, 64, 4, 36);
  prepack_w<<<(4 * 18 * 512 + 255) / 256, 256, 0, stream>>>(w2, wp2, 64, 64, 4, 18);
  prepack_w<<<(4 * 18 * 512 + 255) / 256, 256, 0, stream>>>(w3, wp3, 64, 64, 4, 18);
  prepack_w<<<(2 * 18 * 512 + 255) / 256, 256, 0, stream>>>(woff, wpo, 64, 27, 2, 18);
  prepack_w<<<(4 * 18 * 512 + 255) / 256, 256, 0, stream>>>(wdcn, wpd, 64, 64, 4, 18);
  transpose_bf16<<<(int)((FEAT + 255) / 256), 256, 0, stream>>>(nbr, nbrT);

  dim3 grid(WW / 64, HH, NB);
  // conv1: concat(nbr, ref) f32 NCHW [128ch] -> 64ch NHWC bf16, lrelu
  conv3x3_wmma<128, 64, 4, true, false>
      <<<grid, 128, 0, stream>>>(nbr, ref, wp1, b1, actA, nullptr, nullptr);
  // conv2, conv3: 64 -> 64 NHWC bf16, lrelu (async LDS staging)
  conv3x3_wmma<64, 64, 4, false, false>
      <<<grid, 128, 0, stream>>>(actA, nullptr, wp2, b2, actB, nullptr, nullptr);
  conv3x3_wmma<64, 64, 4, false, false>
      <<<grid, 128, 0, stream>>>(actB, nullptr, wp3, b3, actA, nullptr, nullptr);
  // offset/mask conv: 64 -> 27, fused 15*tanh / sigmoid epilogue into d_out
  conv3x3_wmma<64, 27, 2, false, true>
      <<<grid, 128, 0, stream>>>(actA, nullptr, wpo, boff, nullptr, out_off, out_mask);
  // modulated deformable conv -> feat (lrelu), reads offset/mask from d_out
  dcn_wmma<<<grid, 128, 0, stream>>>(nbrT, out_off, out_mask, wpd, bdcn, out_feat);
}
